// GraphAttentionLayer_18236431138977
// MI455X (gfx1250) — matrix operand order repaired and checked by reading, not compiled, not run on hardware
//
#include <hip/hip_runtime.h>
#include <hip/hip_bf16.h>

#define N_NODES 8192
#define IN_F    256
#define OUT_F   128
#define ALPHA_LRELU 0.2f

typedef __attribute__((ext_vector_type(16))) _Float16 v16h;
typedef __attribute__((ext_vector_type(8)))  float    v8f;
typedef __attribute__((ext_vector_type(4)))  int      i4;
typedef __attribute__((ext_vector_type(4)))  float    f4;
typedef __attribute__((ext_vector_type(4)))  unsigned u4;

union V16U { v16h v; u4 q[2]; };
union Pk8  { _Float16 h[8]; u4 q; };

__device__ inline v8f vzero8() {
  v8f z;
#pragma unroll
  for (int i = 0; i < 8; ++i) z[i] = 0.0f;
  return z;
}

// ---------------------------------------------------------------------------
// Kernel A: Wh = h @ W  (WMMA f16, f32 accum), plus Wh1 = Wh@a[:128],
// Wh2 = Wh@a[128:], plus Wh stored as f16 pre-swizzled into WMMA B-operand
// layout ("whB"): chunk index ((ktile*16 + j)*32 + lane) holds 8 halves,
// flat half f = j*8+p maps to  K = ((f&15)<8 ? hi*8+(f&15) : 16+hi*8+(f&15)-8), feat = (f>>4)*16 + (lane&15)
// where ktile = 32-column group of attention (= 32 rows of Wh).
// One wave per 32 rows of Wh.
// ---------------------------------------------------------------------------
__global__ __launch_bounds__(32)
void wh_kernel(const float* __restrict__ h, const float* __restrict__ W,
               const float* __restrict__ a, u4* __restrict__ whB,
               float* __restrict__ wh1, float* __restrict__ wh2)
{
  __shared__ float lds[32 * OUT_F];          // 16 KB: 32 rows x 128 feats
  const int lane  = threadIdx.x & 31;
  const int mrow  = lane & 15;
  const int hi    = lane >> 4;               // 0 or 1
  const int rbase = blockIdx.x * 32;

  v8f acc[2][8];
#pragma unroll
  for (int mt = 0; mt < 2; ++mt)
#pragma unroll
    for (int n = 0; n < 8; ++n) acc[mt][n] = vzero8();

  const int kb = hi * 8;                     // A-layout K base per half-wave
  for (int ks = 0; ks < IN_F; ks += 32) {
    // A operand (16x32 f16): lane holds M = lane&15, halves 0..7 -> K=kb+i,
    // halves 8..15 -> K=16+kb+i
    v16h Aop[2];
#pragma unroll
    for (int mt = 0; mt < 2; ++mt) {
      const float* hr = h + (size_t)(rbase + mt * 16 + mrow) * IN_F + ks;
#pragma unroll
      for (int i = 0; i < 8; ++i) Aop[mt][i]     = (_Float16)hr[kb + i];
#pragma unroll
      for (int i = 0; i < 8; ++i) Aop[mt][8 + i] = (_Float16)hr[16 + kb + i];
    }
    // B operand (32x16 f16): lane holds N = lane&15, half i<8 -> K = kb+i, i>=8 -> K = 16+kb+(i-8)
    const int kb2 = hi * 16;
#pragma unroll
    for (int n = 0; n < 8; ++n) {
      v16h Bop;
      const int ncol = n * 16 + mrow;
#pragma unroll
      for (int i = 0; i < 16; ++i)
        Bop[i] = (_Float16)W[(size_t)(ks + (i < 8 ? kb + i : 16 + kb + (i - 8))) * OUT_F + ncol];
      acc[0][n] = __builtin_amdgcn_wmma_f32_16x16x32_f16(
          false, Aop[0], false, Bop, (short)0, acc[0][n], false, false);
      acc[1][n] = __builtin_amdgcn_wmma_f32_16x16x32_f16(
          false, Aop[1], false, Bop, (short)0, acc[1][n], false, false);
    }
  }

  // C layout: lane holds rows (hi*8 + r), cols n*16 + (lane&15) -> stage in LDS
#pragma unroll
  for (int mt = 0; mt < 2; ++mt)
#pragma unroll
    for (int n = 0; n < 8; ++n)
#pragma unroll
      for (int r = 0; r < 8; ++r)
        lds[(mt * 16 + hi * 8 + r) * OUT_F + n * 16 + mrow] = acc[mt][n][r];
  __syncthreads();

  // Wh1 / Wh2: one row per lane (32 rows per block)
  {
    const float* row = &lds[lane * OUT_F];
    float s1 = 0.0f, s2 = 0.0f;
#pragma unroll 8
    for (int f = 0; f < OUT_F; ++f) {
      s1 += row[f] * a[f];
      s2 += row[f] * a[OUT_F + f];
    }
    wh1[rbase + lane] = s1;
    wh2[rbase + lane] = s2;
  }

  // whB store: this block is exactly one ktile (= 32 Wh rows)
  const size_t kt = (size_t)blockIdx.x;
#pragma unroll
  for (int j = 0; j < 16; ++j) {
    Pk8 pk;
#pragma unroll
    for (int p = 0; p < 8; ++p) {
      const int f  = j * 8 + p;
      const int i  = f & 15;        // K within half-wave group
      const int nt = f >> 4;        // feature tile
      pk.h[p] = (_Float16)lds[(i < 8 ? hi * 8 + i : 16 + hi * 8 + (i - 8)) * OUT_F + nt * 16 + mrow];
    }
    whB[(kt * 16 + j) * 32 + lane] = pk.q;
  }
}

// ---------------------------------------------------------------------------
// Kernel B: fused masked attention + online softmax + P@Wh (WMMA) + ELU.
// Block = 128 threads (4 waves), 16 attention rows per block.
// Each wave streams a 2048-column strip with private (m, l, acc[16x128]),
// merged at the end through LDS.
// ---------------------------------------------------------------------------
__global__ __launch_bounds__(128)
void att_kernel(const float* __restrict__ wh1v, const float* __restrict__ wh2v,
                const u4* __restrict__ whB,
                const int* __restrict__ adj, const float* __restrict__ lpred,
                const float* __restrict__ p_eta1, const float* __restrict__ p_eta2,
                const float* __restrict__ p_eta3, const float* __restrict__ p_th1,
                const float* __restrict__ p_th2, float* __restrict__ out)
{
  __shared__ float sm[4][16];
  __shared__ float sl[4][16];
  __shared__ float sacc[4][16][OUT_F];       // 32 KB

  const int tid  = threadIdx.x;
  const int lane = tid & 31;
  const int w    = tid >> 5;                 // wave id 0..3
  const int mrow = lane & 15;
  const int hi   = lane >> 4;
  const int kb   = hi * 8;
  const int row  = blockIdx.x * 16 + mrow;
  const size_t rowoff = (size_t)row * N_NODES;

  const float eta1 = *p_eta1, eta2 = *p_eta2, eta3 = *p_eta3;
  const float th1  = *p_th1,  th2  = *p_th2;
  const float wh1r = wh1v[row];

  float m_run = -INFINITY;
  float l_run = 0.0f;
  v8f acc[8];
#pragma unroll
  for (int n = 0; n < 8; ++n) acc[n] = vzero8();

  const int c0 = w * (N_NODES / 4);
  const int c1 = c0 + (N_NODES / 4);

  for (int ct = c0; ct < c1; ct += 32) {
    const int*   ap = adj   + rowoff + ct;
    const float* pp = lpred + rowoff + ct;
    // deep prefetch: 4 K-tiles (one 128B line per row per stream) ahead.
    // Speculative on gfx1250 -> silently dropped if the address is bad.
    __builtin_prefetch(ap + 128, 0, 1);
    __builtin_prefetch(pp + 128, 0, 1);

    // ---- issue this tile's loads up front so the clause scheduler can
    // overlap them with the score/softmax VALU work ----
    const i4 A0 = __builtin_nontemporal_load((const i4*)(ap + kb));
    const i4 A1 = __builtin_nontemporal_load((const i4*)(ap + kb + 4));
    const i4 A2 = __builtin_nontemporal_load((const i4*)(ap + 16 + kb));
    const i4 A3 = __builtin_nontemporal_load((const i4*)(ap + 16 + kb + 4));
    const f4 L0 = __builtin_nontemporal_load((const f4*)(pp + kb));
    const f4 L1 = __builtin_nontemporal_load((const f4*)(pp + kb + 4));
    const f4 L2 = __builtin_nontemporal_load((const f4*)(pp + 16 + kb));
    const f4 L3 = __builtin_nontemporal_load((const f4*)(pp + 16 + kb + 4));

    // B operand: 16 coalesced b128 loads (pre-swizzled f16 Wh, L2-resident)
    const size_t kt = (size_t)(ct >> 5);
    u4 bchunk[16];
#pragma unroll
    for (int j = 0; j < 16; ++j)
      bchunk[j] = whB[(kt * 16 + j) * 32 + lane];

    const f4 W0 = *(const f4*)(wh2v + ct + kb);
    const f4 W1 = *(const f4*)(wh2v + ct + kb + 4);
    const f4 W2 = *(const f4*)(wh2v + ct + 16 + kb);
    const f4 W3 = *(const f4*)(wh2v + ct + 16 + kb + 4);

    // 16 masked scores per lane, laid out exactly as the WMMA A operand:
    // half i<8 -> col = ct + kb + i ; half i>=8 -> col = ct + 16 + kb + (i-8)
    float s[16];
#pragma unroll
    for (int p = 0; p < 4; ++p) {
      auto sc = [&](int av, float lv, float w2) -> float {
        float e = wh1r + w2;
        e = (e > 0.0f) ? e : ALPHA_LRELU * e;
        float lp = (av == 0) ? 0.0f : lv;
        return (lp <= th1) ? eta1 : ((lp < th2) ? eta2 * e : eta3 * e);
      };
      s[p]      = sc(A0[p], L0[p], W0[p]);
      s[4 + p]  = sc(A1[p], L1[p], W1[p]);
      s[8 + p]  = sc(A2[p], L2[p], W2[p]);
      s[12 + p] = sc(A3[p], L3[p], W3[p]);
    }

    // online softmax: row max over the 32-col tile (combine lane pair l, l^16)
    float tmax = s[0];
#pragma unroll
    for (int i = 1; i < 16; ++i) tmax = fmaxf(tmax, s[i]);
    tmax = fmaxf(tmax, __shfl_xor(tmax, 16, 32));
    const float m_new = fmaxf(m_run, tmax);
    const float scale = __expf(m_run - m_new);   // == 1.0 when max unchanged
    const bool  grew  = (m_new > m_run);

    v16h pA;
    float tsum = 0.0f;
#pragma unroll
    for (int i = 0; i < 16; ++i) {
      const float pv = __expf(s[i] - m_new);
      tsum += pv;
      pA[i] = (_Float16)pv;
    }
    tsum += __shfl_xor(tsum, 16, 32);
    l_run = l_run * scale + tsum;
    m_run = m_new;

    // rescale accumulator only when some row max actually changed (wave-
    // uniform branch; scale is exactly 1.0 for unchanged rows).
    // C-layout lane holds rows hi*8 + r; scale for row q lives in lane q.
    if (__any(grew)) {
#pragma unroll
      for (int r = 0; r < 8; ++r) {
        const float sr = __shfl(scale, hi * 8 + r, 32);
#pragma unroll
        for (int n = 0; n < 8; ++n) acc[n][r] *= sr;
      }
    }

#pragma unroll
    for (int n = 0; n < 8; ++n) {
      V16U bb;
      bb.q[0] = bchunk[2 * n];
      bb.q[1] = bchunk[2 * n + 1];
      acc[n] = __builtin_amdgcn_wmma_f32_16x16x32_f16(
          false, pA, false, bb.v, (short)0, acc[n], false, false);
    }
  }

  // ---- cross-wave softmax merge via LDS ----
  if (lane < 16) { sm[w][lane] = m_run; sl[w][lane] = l_run; }
#pragma unroll
  for (int n = 0; n < 8; ++n)
#pragma unroll
    for (int r = 0; r < 8; ++r)
      sacc[w][hi * 8 + r][n * 16 + mrow] = acc[n][r];
  __syncthreads();

  // each thread finalizes one row, 16 features
  const int orow  = tid >> 3;
  const int fbase = (tid & 7) * 16;
  float mstar = sm[0][orow];
#pragma unroll
  for (int q = 1; q < 4; ++q) mstar = fmaxf(mstar, sm[q][orow]);
  float coef[4], lstar = 0.0f;
#pragma unroll
  for (int q = 0; q < 4; ++q) {
    coef[q] = __expf(sm[q][orow] - mstar);
    lstar  += sl[q][orow] * coef[q];
  }
  const float inv = 1.0f / lstar;
  const size_t obase = (size_t)(blockIdx.x * 16 + orow) * OUT_F;
#pragma unroll
  for (int t = 0; t < 16; ++t) {
    const int f = fbase + t;
    float v = 0.0f;
#pragma unroll
    for (int q = 0; q < 4; ++q) v += sacc[q][orow][f] * coef[q];
    v *= inv;
    v = (v > 0.0f) ? v : (__expf(v) - 1.0f);   // ELU (alpha = 1)
    out[obase + f] = v;
  }
}

// ---------------------------------------------------------------------------
extern "C" void kernel_launch(void* const* d_in, const int* in_sizes, int n_in,
                              void* d_out, int out_size, void* d_ws, size_t ws_size,
                              hipStream_t stream) {
  const float* h     = (const float*)d_in[0];
  const float* W     = (const float*)d_in[1];
  const float* a     = (const float*)d_in[2];
  const float* eta1  = (const float*)d_in[3];
  const float* eta2  = (const float*)d_in[4];
  const float* eta3  = (const float*)d_in[5];
  const float* lpred = (const float*)d_in[6];
  const float* th1   = (const float*)d_in[7];
  const float* th2   = (const float*)d_in[8];
  const int*   adj   = (const int*)d_in[9];

  char* ws = (char*)d_ws;
  u4*    whB = (u4*)ws;                                        // 2 MB f16 Wh (B-layout)
  float* wh1 = (float*)(ws + (size_t)N_NODES * OUT_F * 2);     // 32 KB
  float* wh2 = wh1 + N_NODES;                                  // 32 KB

  wh_kernel<<<N_NODES / 32, 32, 0, stream>>>(h, W, a, whB, wh1, wh2);
  att_kernel<<<N_NODES / 16, 128, 0, stream>>>(wh1, wh2, whB, adj, lpred,
                                               eta1, eta2, eta3, th1, th2,
                                               (float*)d_out);
}
